// DalleBartDecoder_11089605559112
// MI455X (gfx1250) — compile-verified, matrix operand order checked by reading, not run on hardware
//
#include <hip/hip_runtime.h>
#include <hip/hip_bf16.h>
#include <math.h>

// Problem constants (reference: L,B,S_IMG,S_TXT,E,H,G,V = 12,2,256,64,1024,16,2730,16385)
#define LNUM  12
#define BNUM  2
#define SIMG  256
#define STXT  64
#define EDIM  1024
#define HNUM  16
#define DHEAD 64
#define GDIM  2730
#define GPAD  2752   // GDIM rounded up to 32 (K-padding for fc2 contraction)
#define VNUM  16385
#define KCHUNK 128   // split-K chunk (4 WMMA k-steps)

typedef __attribute__((ext_vector_type(16))) _Float16 v16h;
typedef __attribute__((ext_vector_type(8)))  float    v8f;

static __device__ __forceinline__ v8f wmma_step(const v16h& a, const v16h& b, v8f acc) {
    return __builtin_amdgcn_wmma_f32_16x16x32_f16(false, a, false, b, (short)0, acc, false, false);
}

static __device__ __forceinline__ void load_a_frag(const float* __restrict__ Arow, int k0, v16h& a) {
    const float4 a0 = *reinterpret_cast<const float4*>(Arow + k0);
    const float4 a1 = *reinterpret_cast<const float4*>(Arow + k0 + 4);
    const float4 a2 = *reinterpret_cast<const float4*>(Arow + k0 + 16);
    const float4 a3 = *reinterpret_cast<const float4*>(Arow + k0 + 20);
    a[0]  = (_Float16)a0.x; a[1]  = (_Float16)a0.y; a[2]  = (_Float16)a0.z; a[3]  = (_Float16)a0.w;
    a[4]  = (_Float16)a1.x; a[5]  = (_Float16)a1.y; a[6]  = (_Float16)a1.z; a[7]  = (_Float16)a1.w;
    a[8]  = (_Float16)a2.x; a[9]  = (_Float16)a2.y; a[10] = (_Float16)a2.z; a[11] = (_Float16)a2.w;
    a[12] = (_Float16)a3.x; a[13] = (_Float16)a3.y; a[14] = (_Float16)a3.z; a[15] = (_Float16)a3.w;
}

// ---------------------------------------------------------------------------
// Split-K WMMA GEMM partial: part[z, 0:Mpad, 0:N] = A[M,K] @ W[K,N] over the
// z-th K-chunk. fp32 data, f16 convert in-register, f32 accumulate.
// One wave per (32-col, 16-row, k-chunk) tile; 2 WMMAs per k-step.
// use_async (N%4==0): DOUBLE-BUFFERED staging of 32x32 weight tiles into LDS
// via global_load_async_to_lds_b128 (8 x 512B coalesced async transfers per
// tile). Async loads complete in-order on ASYNCcnt, so s_wait_asynccnt 0x8
// after issuing the next batch guarantees the current tile has landed while
// the next transfer overlaps this step's ds reads + cvts + WMMAs.
// Otherwise: direct strided b32 loads through strength-reduced row pointers.
// K tail (K%32 != 0): A is zero-padded to lda (host guarantees), W k-row
// index clamped -> duplicated finite rows x zero A contribute exactly 0.
// ---------------------------------------------------------------------------
__global__ void wmma_gemm(const float* __restrict__ A, int lda,
                          const float* __restrict__ W,
                          float* __restrict__ part,
                          int M, int N, int K, int use_async)
{
    __shared__ float tile[2 * 32 * 32];   // 2 x 4KB staging tiles (async path)
    const int lane = threadIdx.x & 31;
    const int half = lane >> 4;
    const int l15  = lane & 15;
    const int Mpad = gridDim.y * 16;

    const int msafe = min(blockIdx.y * 16 + l15, M - 1);
    const int n0 = blockIdx.x * 32 + l15;
    const int ns0 = min(n0, N - 1);
    const int ns1 = min(n0 + 16, N - 1);
    const int dcol = ns1 - ns0;
    const float* __restrict__ Arow = A + (size_t)msafe * lda + half * 8;
    const float* __restrict__ Wc0  = W + ns0;

    const int kbeg = blockIdx.z * KCHUNK;
    const int kend = min(kbeg + KCHUNK, K);

    v8f acc0 = {}, acc1 = {};

    if (use_async) {
        // per-lane async source/dest mapping: instr j covers rows j*4+(lane>>3),
        // 16B column chunk (lane&7)*4 of the 32-wide tile
        const int rsub   = lane >> 3;
        const int cchunk = (lane & 7) * 4;
        const int cbase  = min(blockIdx.x * 32 + cchunk, N - 4);   // 16B aligned (N%4==0)
        const unsigned ldsrow = (unsigned)((rsub * 32 + cchunk) * 4);
        const unsigned tbase  = (unsigned)(size_t)tile;

        auto issue_tile = [&](int k0, int buf) {
#pragma unroll
            for (int j = 0; j < 8; ++j) {
                const int rs = min(k0 + j * 4 + rsub, K - 1);      // row clamp (tail)
                const unsigned long long ga =
                    (unsigned long long)(size_t)(W + (size_t)rs * N + cbase);
                const unsigned ld = tbase + (unsigned)(buf * 4096) + ldsrow + (unsigned)(j * 512);
                asm volatile("global_load_async_to_lds_b128 %0, %1, off"
                             :: "v"(ld), "v"(ga) : "memory");
            }
        };
        auto consume_tile = [&](int k0, int buf) {
            v16h a;
            load_a_frag(Arow, k0, a);
            v16h b0, b1;
            const int base = buf * 1024 + half * 8 * 32 + l15;
#pragma unroll
            for (int i = 0; i < 8; ++i) {
                b0[i]     = (_Float16)tile[base + i * 32];
                b0[i + 8] = (_Float16)tile[base + (16 + i) * 32];
                b1[i]     = (_Float16)tile[base + i * 32 + 16];
                b1[i + 8] = (_Float16)tile[base + (16 + i) * 32 + 16];
            }
            acc0 = wmma_step(a, b0, acc0);
            acc1 = wmma_step(a, b1, acc1);
        };

        int cur = 0;
        issue_tile(kbeg, 0);
        int k0 = kbeg;
        for (; k0 + 32 < kend; k0 += 32) {
            issue_tile(k0 + 32, cur ^ 1);                          // prefetch next tile
            asm volatile("s_wait_asynccnt 0x8" ::: "memory");      // current tile landed
            consume_tile(k0, cur);
            cur ^= 1;
        }
        asm volatile("s_wait_asynccnt 0x0" ::: "memory");          // drain last tile
        consume_tile(k0, cur);
    } else {
        // strength-reduced row pointers: one 64-bit add per pointer per k-step
        const float* pr[8];
        const float* qr[8];
#pragma unroll
        for (int j = 0; j < 8; ++j) {
            pr[j] = Wc0 + (size_t)(kbeg + half * 8 + j) * N;
            qr[j] = pr[j] + (size_t)16 * N;
        }
        const size_t stepN = (size_t)32 * N;
        int k0 = kbeg;
        for (; k0 + 32 <= kend; k0 += 32) {
            v16h a, b0, b1;
            load_a_frag(Arow, k0, a);
#pragma unroll
            for (int i = 0; i < 8; ++i) {
                b0[i]     = (_Float16)pr[i][0];
                b0[i + 8] = (_Float16)qr[i][0];
                b1[i]     = (_Float16)pr[i][dcol];
                b1[i + 8] = (_Float16)qr[i][dcol];
            }
            if (k0 + 32 < K)
                __builtin_prefetch(pr[0] + stepN, 0, 1);           // global_prefetch_b8
            acc0 = wmma_step(a, b0, acc0);
            acc1 = wmma_step(a, b1, acc1);
#pragma unroll
            for (int j = 0; j < 8; ++j) { pr[j] += stepN; qr[j] += stepN; }
        }
        if (k0 < kend) {   // K tail: clamp W k-rows, A zero-padded past K
            const int kb = k0 + half * 8;
            v16h a, b0, b1;
            load_a_frag(Arow, k0, a);
#pragma unroll
            for (int i = 0; i < 8; ++i) {
                const size_t r1 = (size_t)min(kb + i,      K - 1) * N;
                const size_t r2 = (size_t)min(kb + 16 + i, K - 1) * N;
                b0[i] = (_Float16)Wc0[r1]; b0[i + 8] = (_Float16)Wc0[r2];
                b1[i] = (_Float16)Wc0[r1 + dcol]; b1[i + 8] = (_Float16)Wc0[r2 + dcol];
            }
            acc0 = wmma_step(a, b0, acc0);
            acc1 = wmma_step(a, b1, acc1);
        }
    }

    // store partial tile: rows always (garbage rows never reduced), cols guarded
    float* __restrict__ pz = part + (size_t)blockIdx.z * Mpad * N;
#pragma unroll
    for (int i = 0; i < 8; ++i) {
        const int mm = blockIdx.y * 16 + half * 8 + i;
        const int nnA = blockIdx.x * 32 + l15;
        const int nnB = nnA + 16;
        if (nnA < N) pz[(size_t)mm * N + nnA] = acc0[i];
        if (nnB < N) pz[(size_t)mm * N + nnB] = acc1[i];
    }
}

// Deterministic split-K reduction: C[m,n] = sum_z part[z,m,n] (+ resid).
__global__ void splitk_reduce(const float* __restrict__ part, const float* __restrict__ resid,
                              float* __restrict__ C, int Mpad, int N, int M, int Z)
{
    const int idx = blockIdx.x * blockDim.x + threadIdx.x;
    if (idx >= M * N) return;
    const int m = idx / N, n = idx % N;
    float s = 0.f;
    for (int z = 0; z < Z; ++z) s += part[((size_t)z * Mpad + m) * N + n];
    if (resid) s += resid[idx];
    C[idx] = s;
}

// ---------------------------------------------------------------------------
// LayerNorm over last dim n, one block per row, fused residual, output stride.
// ---------------------------------------------------------------------------
__global__ void ln_kernel(const float* __restrict__ in, const float* __restrict__ w,
                          const float* __restrict__ bb, const float* __restrict__ resid,
                          float* __restrict__ out, int n, int ostride)
{
    __shared__ float red[256];
    const int row = blockIdx.x;
    const int tid = threadIdx.x;
    const float* x = in + (size_t)row * n;

    float s = 0.f;
    for (int i = tid; i < n; i += blockDim.x) s += x[i];
    red[tid] = s; __syncthreads();
    for (int off = 128; off > 0; off >>= 1) { if (tid < off) red[tid] += red[tid + off]; __syncthreads(); }
    const float mu = red[0] / (float)n;
    __syncthreads();

    float v = 0.f;
    for (int i = tid; i < n; i += blockDim.x) { float d = x[i] - mu; v += d * d; }
    red[tid] = v; __syncthreads();
    for (int off = 128; off > 0; off >>= 1) { if (tid < off) red[tid] += red[tid + off]; __syncthreads(); }
    const float rstd = rsqrtf(red[0] / (float)n + 1e-5f);

    for (int i = tid; i < n; i += blockDim.x) {
        float y = (x[i] - mu) * rstd * w[i] + bb[i];
        if (resid) y += resid[(size_t)row * n + i];
        out[(size_t)row * ostride + i] = y;
    }
}

// ---------------------------------------------------------------------------
// Single-query multi-head attention. One block per (b,h).
// ---------------------------------------------------------------------------
__global__ void attend_kernel(const float* __restrict__ q,
                              const float* __restrict__ Kc, int kb_stride,
                              const float* __restrict__ Vc, int vb_stride,
                              const unsigned char* __restrict__ mask, int mask_stride,
                              const int* __restrict__ tokp, int Smax,
                              float* __restrict__ out)
{
    __shared__ float sc[SIMG];
    __shared__ float qs[DHEAD];
    __shared__ float red[256];
    __shared__ float smax_s, ssum_s;
    const int tid = threadIdx.x;
    const int b = blockIdx.x / HNUM;
    const int h = blockIdx.x % HNUM;

    int S = Smax;
    if (tokp) { S = *tokp + 1; if (S > Smax) S = Smax; if (S < 1) S = 1; }

    if (tid < DHEAD) qs[tid] = q[(size_t)b * EDIM + h * DHEAD + tid];
    __syncthreads();

    float sco = -3.4e38f;
    if (tid < S) {
        const float* kr = Kc + (size_t)b * kb_stride + (size_t)tid * EDIM + h * DHEAD;
        float d = 0.f;
#pragma unroll 8
        for (int c = 0; c < DHEAD; ++c) d += qs[c] * kr[c];
        sco = d * 0.125f;  // 1/sqrt(64)
        if (mask && !mask[b * mask_stride + tid]) sco -= 1e12f;
        sc[tid] = sco;
    }
    red[tid] = sco; __syncthreads();
    for (int off = 128; off > 0; off >>= 1) { if (tid < off) red[tid] = fmaxf(red[tid], red[tid + off]); __syncthreads(); }
    if (tid == 0) smax_s = red[0];
    __syncthreads();

    float e = 0.f;
    if (tid < S) { e = expf(sc[tid] - smax_s); sc[tid] = e; }
    red[tid] = e; __syncthreads();
    for (int off = 128; off > 0; off >>= 1) { if (tid < off) red[tid] += red[tid + off]; __syncthreads(); }
    if (tid == 0) ssum_s = red[0];
    __syncthreads();

    if (tid < DHEAD) {
        float o = 0.f;
        const float* vbp = Vc + (size_t)b * vb_stride + h * DHEAD + tid;
        for (int j = 0; j < S; ++j) o += sc[j] * vbp[(size_t)j * EDIM];
        out[(size_t)b * EDIM + h * DHEAD + tid] = o / ssum_s;
    }
}

__global__ void kv_write(const float* __restrict__ k, const float* __restrict__ v,
                         float* __restrict__ astate_l, const int* __restrict__ tokp)
{
    const int tid = blockIdx.x * blockDim.x + threadIdx.x;
    if (tid >= BNUM * EDIM) return;
    const int b = tid / EDIM, e = tid % EDIM;
    const int tok = *tokp;
    astate_l[((size_t)b * SIMG + tok) * EDIM + e]          = k[tid];
    astate_l[((size_t)(b + BNUM) * SIMG + tok) * EDIM + e] = v[tid];
}

__global__ void embed_kernel(const float* __restrict__ etok, const float* __restrict__ epos,
                             const int* __restrict__ prev, const int* __restrict__ tokp,
                             float* __restrict__ dec)
{
    const int tid = blockIdx.x * blockDim.x + threadIdx.x;
    if (tid >= BNUM * EDIM) return;
    const int e = tid % EDIM;
    dec[tid] = etok[(size_t)prev[0] * EDIM + e] + epos[(size_t)(*tokp) * EDIM + e];
}

__global__ void gelu_mul(const float* __restrict__ a, const float* __restrict__ b,
                         float* __restrict__ o, int n)
{
    const int i = blockIdx.x * blockDim.x + threadIdx.x;
    if (i < n) {
        float x = a[i];
        float g = 0.5f * x * (1.f + erff(x * 0.70710678118654752f));  // exact gelu
        o[i] = g * b[i];
    }
}

// zero the K-padding columns of zg (rows have stride GPAD, valid cols < GDIM)
__global__ void zgpad_kernel(float* __restrict__ zg)
{
    const int tid = threadIdx.x;                    // 64 threads
    const int row = tid / (GPAD - GDIM);
    const int col = GDIM + tid % (GPAD - GDIM);
    if (row < BNUM) zg[(size_t)row * GPAD + col] = 0.f;
}

__global__ void cfg_combine(const float* __restrict__ logits, float* __restrict__ lcomb,
                            float* __restrict__ scratch)
{
    const int v = blockIdx.x * blockDim.x + threadIdx.x;
    if (v < VNUM) {
        float l = logits[v] * (1.f - 10.f) + logits[VNUM + v] * 10.f;
        lcomb[v] = l; scratch[v] = l;
    }
}

__global__ void topk50(float* __restrict__ scratch, float* __restrict__ thr)
{
    __shared__ float sv[1024];
    __shared__ int   si[1024];
    const int tid = threadIdx.x;
    for (int it = 0; it < 50; ++it) {
        float best = -3.4e38f; int bi = 0;
        for (int v = tid; v < VNUM; v += 1024) {
            float x = scratch[v];
            if (x > best) { best = x; bi = v; }
        }
        sv[tid] = best; si[tid] = bi; __syncthreads();
        for (int off = 512; off > 0; off >>= 1) {
            if (tid < off) {
                if (sv[tid + off] > sv[tid] ||
                    (sv[tid + off] == sv[tid] && si[tid + off] < si[tid])) {
                    sv[tid] = sv[tid + off]; si[tid] = si[tid + off];
                }
            }
            __syncthreads();
        }
        if (tid == 0) {
            if (it == 0)  thr[0] = sv[0];
            if (it == 49) thr[1] = sv[0];
            scratch[si[0]] = -3.4e38f;  // remove one instance (duplicate-safe)
        }
        __syncthreads();
    }
}

__global__ void probs_kernel(const float* __restrict__ lcomb, const float* __restrict__ thr,
                             float* __restrict__ out)
{
    const int v = blockIdx.x * blockDim.x + threadIdx.x;
    if (v < VNUM) {
        float l = lcomb[v];
        out[v] = (l < thr[1]) ? 0.f : expf(l - thr[0]);
    }
}

// ---------------------------------------------------------------------------
static inline void gemm(const float* A, int lda, const float* W, const float* resid,
                        float* C, int M, int N, int K, float* part, hipStream_t st)
{
    const int Z  = (K + KCHUNK - 1) / KCHUNK;
    const int Mt = (M + 15) / 16;
    const int use_async = (N % 4) == 0;   // b128 async-to-LDS needs 16B alignment
    dim3 g((N + 31) / 32, Mt, Z);
    wmma_gemm<<<g, 32, 0, st>>>(A, lda, W, part, M, N, K, use_async);
    const int tot = M * N;
    splitk_reduce<<<(tot + 255) / 256, 256, 0, st>>>(part, resid, C, Mt * 16, N, M, Z);
}

extern "C" void kernel_launch(void* const* d_in, const int* in_sizes, int n_in,
                              void* d_out, int out_size, void* d_ws, size_t ws_size,
                              hipStream_t stream)
{
    (void)in_sizes; (void)n_in; (void)out_size; (void)ws_size;
    const unsigned char* am  = (const unsigned char*)d_in[0];   // [B,STXT] bool
    const float* enc         = (const float*)d_in[1];           // [B,STXT,E]
    const float* astate_in   = (const float*)d_in[2];           // [L,2B,SIMG,E]
    const int*   prev        = (const int*)d_in[3];
    const int*   tokp        = (const int*)d_in[4];
    const float* etok        = (const float*)d_in[5];
    const float* epos        = (const float*)d_in[6];
    const float* ln_emb_w    = (const float*)d_in[7];
    const float* ln_emb_b    = (const float*)d_in[8];
    const float* Wqs = (const float*)d_in[9];  const float* Wks = (const float*)d_in[10];
    const float* Wvs = (const float*)d_in[11]; const float* Wos = (const float*)d_in[12];
    const float* Wqc = (const float*)d_in[13]; const float* Wkc = (const float*)d_in[14];
    const float* Wvc = (const float*)d_in[15]; const float* Woc = (const float*)d_in[16];
    const float* ln1_w = (const float*)d_in[17]; const float* ln1_b = (const float*)d_in[18];
    const float* ln2_w = (const float*)d_in[19]; const float* ln2_b = (const float*)d_in[20];
    const float* ln3_w = (const float*)d_in[21]; const float* ln3_b = (const float*)d_in[22];
    const float* ln4_w = (const float*)d_in[23]; const float* ln4_b = (const float*)d_in[24];
    const float* g0w   = (const float*)d_in[25]; const float* g0b   = (const float*)d_in[26];
    const float* gf0   = (const float*)d_in[27]; const float* gf1   = (const float*)d_in[28];
    const float* g1w   = (const float*)d_in[29]; const float* g1b   = (const float*)d_in[30];
    const float* gf2   = (const float*)d_in[31];
    const float* fin_w = (const float*)d_in[32]; const float* fin_b = (const float*)d_in[33];
    const float* lm    = (const float*)d_in[34];

    float* out_probs  = (float*)d_out;            // [VNUM]
    float* out_astate = out_probs + VNUM;         // [L,2B,SIMG,E]

    float* ws   = (float*)d_ws;
    float* dec  = ws;                   // [2,E]
    float* x    = ws + 2048;            // [2,E]
    float* q    = ws + 4096;            // [2,E]
    float* kb   = ws + 6144;            // [2,E]
    float* vb   = ws + 8192;            // [2,E]
    float* at   = ws + 10240;           // [2,E]
    float* xo   = ws + 12288;           // [2,E]
    float* kenc = ws + 14336;           // [2,64,E]
    float* venc = kenc + (size_t)BNUM * STXT * EDIM;
    float* h0   = venc + (size_t)BNUM * STXT * EDIM;   // [2,G]
    float* h1   = h0 + (size_t)BNUM * GDIM;
    float* hg   = h1 + (size_t)BNUM * GDIM;
    float* zg   = hg + (size_t)BNUM * GDIM;            // [2,GPAD] (zero-padded K)
    float* logits  = zg + (size_t)BNUM * GPAD;         // [2,V]
    float* lcomb   = logits + (size_t)BNUM * VNUM;     // [V]
    float* scratch = lcomb + VNUM;                     // [V]
    float* thr     = scratch + VNUM;                   // [2]
    float* part    = thr + 16;                         // split-K partials

    hipMemcpyAsync(out_astate, astate_in,
                   (size_t)LNUM * 2 * BNUM * SIMG * EDIM * sizeof(float),
                   hipMemcpyDeviceToDevice, stream);

    zgpad_kernel<<<1, 64, 0, stream>>>(zg);
    embed_kernel<<<8, 256, 0, stream>>>(etok, epos, prev, tokp, dec);
    ln_kernel<<<BNUM, 256, 0, stream>>>(dec, ln_emb_w, ln_emb_b, nullptr, dec, EDIM, EDIM);

    for (int l = 0; l < LNUM; ++l) {
        const size_t ee = (size_t)EDIM * EDIM;
        const float* wq = Wqs + (size_t)l * ee;
        const float* wk = Wks + (size_t)l * ee;
        const float* wv = Wvs + (size_t)l * ee;
        const float* wo = Wos + (size_t)l * ee;
        const float* cq = Wqc + (size_t)l * ee;
        const float* ck = Wkc + (size_t)l * ee;
        const float* cv = Wvc + (size_t)l * ee;
        const float* co = Woc + (size_t)l * ee;
        float* astl = out_astate + (size_t)l * 2 * BNUM * SIMG * EDIM;

        // ---- self-attention (KV-cache) ----
        ln_kernel<<<BNUM, 256, 0, stream>>>(dec, ln1_w + l * EDIM, ln1_b + l * EDIM, nullptr, x, EDIM, EDIM);
        gemm(x, EDIM, wq, nullptr, q,  BNUM, EDIM, EDIM, part, stream);
        gemm(x, EDIM, wk, nullptr, kb, BNUM, EDIM, EDIM, part + 0x40000, stream);
        gemm(x, EDIM, wv, nullptr, vb, BNUM, EDIM, EDIM, part + 0x80000, stream);
        kv_write<<<8, 256, 0, stream>>>(kb, vb, astl, tokp);
        attend_kernel<<<BNUM * HNUM, 256, 0, stream>>>(
            q, astl, SIMG * EDIM, astl + (size_t)BNUM * SIMG * EDIM, SIMG * EDIM,
            nullptr, 0, tokp, SIMG, at);
        gemm(at, EDIM, wo, nullptr, xo, BNUM, EDIM, EDIM, part, stream);
        ln_kernel<<<BNUM, 256, 0, stream>>>(xo, ln2_w + l * EDIM, ln2_b + l * EDIM, dec, dec, EDIM, EDIM);

        // ---- cross-attention over text encoder state ----
        ln_kernel<<<BNUM, 256, 0, stream>>>(dec, ln3_w + l * EDIM, ln3_b + l * EDIM, nullptr, x, EDIM, EDIM);
        gemm(x, EDIM, cq, nullptr, q, BNUM, EDIM, EDIM, part, stream);
        gemm(enc, EDIM, ck, nullptr, kenc, BNUM * STXT, EDIM, EDIM, part + 0x40000, stream);
        gemm(enc, EDIM, cv, nullptr, venc, BNUM * STXT, EDIM, EDIM, part + 0x140000, stream);
        attend_kernel<<<BNUM * HNUM, 256, 0, stream>>>(
            q, kenc, STXT * EDIM, venc, STXT * EDIM, am, STXT, nullptr, STXT, at);
        gemm(at, EDIM, co, nullptr, xo, BNUM, EDIM, EDIM, part, stream);
        ln_kernel<<<BNUM, 256, 0, stream>>>(xo, ln4_w + l * EDIM, ln4_b + l * EDIM, dec, dec, EDIM, EDIM);

        // ---- GLU FFN ----
        ln_kernel<<<BNUM, 256, 0, stream>>>(dec, g0w + l * EDIM, g0b + l * EDIM, nullptr, x, EDIM, EDIM);
        gemm(x, EDIM, gf0 + (size_t)l * EDIM * GDIM, nullptr, h0, BNUM, GDIM, EDIM, part, stream);
        gemm(x, EDIM, gf1 + (size_t)l * EDIM * GDIM, nullptr, h1, BNUM, GDIM, EDIM, part + 0x80000, stream);
        gelu_mul<<<(BNUM * GDIM + 255) / 256, 256, 0, stream>>>(h0, h1, hg, BNUM * GDIM);
        ln_kernel<<<BNUM, 256, 0, stream>>>(hg, g1w + l * GDIM, g1b + l * GDIM, nullptr, zg, GDIM, GPAD);
        gemm(zg, GPAD, gf2 + (size_t)l * GDIM * EDIM, dec, dec, BNUM, EDIM, GDIM, part, stream);
    }

    // ---- lm head + CFG + top-50 ----
    ln_kernel<<<BNUM, 256, 0, stream>>>(dec, fin_w, fin_b, nullptr, x, EDIM, EDIM);
    gemm(x, EDIM, lm, nullptr, logits, BNUM, VNUM, EDIM, part, stream);
    cfg_combine<<<(VNUM + 255) / 256, 256, 0, stream>>>(logits, lcomb, scratch);
    topk50<<<1, 1024, 0, stream>>>(scratch, thr);
    probs_kernel<<<(VNUM + 255) / 256, 256, 0, stream>>>(lcomb, thr, out_probs);
}
